// LearningToRotateAttentionLayer_40011915329802
// MI455X (gfx1250) — matrix operand order repaired and checked by reading, not compiled, not run on hardware
//
#include <hip/hip_runtime.h>
#include <hip/hip_bf16.h>
#include <math.h>

// Problem dims (fixed by reference)
#define BB 16
#define LL 512
#define DM 512
#define HH 8
#define EE 64
#define RD 256        // rotated feature dim = 2*M*E
#define PI_F 3.14159265358979323846f

typedef __attribute__((ext_vector_type(16))) _Float16 v16h;
typedef __attribute__((ext_vector_type(8)))  _Float16 v8h;
typedef __attribute__((ext_vector_type(4)))  _Float16 v4h;
typedef __attribute__((ext_vector_type(8)))  float    v8f;
typedef __attribute__((ext_vector_type(4)))  unsigned int u32x4;
typedef __attribute__((ext_vector_type(8)))  int      i32x8;
typedef __attribute__((ext_vector_type(4)))  int      i32x4;

// ---------------------------------------------------------------------------
// WMMA helpers (CDNA5 v_wmma_f32_16x16x32_f16, wave32)
// A frag (this code's contiguous k order): lane holds row M=lane&15; half hs=lane>>4; a[j]=A[row][hs*16+j].
// B frag (same k order as A): lane holds col N=lane&15; b[j]=B[hs*16+j][col].
// C/D f32:  c[r] = C[r + 8*hs][lane&15].
// ---------------------------------------------------------------------------
__device__ __forceinline__ v8f wmma_f16(v16h a, v16h b, v8f c) {
  return __builtin_amdgcn_wmma_f32_16x16x32_f16(false, a, false, b, (short)0, c,
                                                false, false);
}

__device__ __forceinline__ v16h afrag_f16(const _Float16* p, int hs) {
  v8h lo = *(const v8h*)(p + hs * 16);
  v8h hi = *(const v8h*)(p + hs * 16 + 8);
  v16h r;
#pragma unroll
  for (int i = 0; i < 8; ++i) { r[i] = lo[i]; r[8 + i] = hi[i]; }
  return r;
}

__device__ __forceinline__ v4h cvt4(float4 x) {
  v4h r;
  r[0] = (_Float16)x.x; r[1] = (_Float16)x.y; r[2] = (_Float16)x.z; r[3] = (_Float16)x.w;
  return r;
}

// ---------------------------------------------------------------------------
// Tensor Data Mover: DMA a 64-row x 128-half tile (f16) from a (64 x 512)
// row-major f16 tensor into LDS, padding each 64-DWORD row with 4 DWORDs
// (-> LDS row stride 136 halves, matching sXt[64][136]).
// D# packing per CDNA5 ISA (group0: count/lds/global/type,
// group1: data_size=2B, pad_enable, pad_interval=64dw, pad_amount=4dw,
// tensor_dim0=512, tensor_dim1=64, tile_dim0=128, tile_dim1=64, stride=512).
// ---------------------------------------------------------------------------
__device__ __forceinline__ void tdm_stage_64x128(_Float16* lds,
                                                 const _Float16* gsrc) {
  u32x4 g0 = {};
  i32x8 g1 = {};
  i32x4 z4 = {};
  unsigned ldsa = (unsigned)(unsigned long long)(size_t)lds;
  unsigned long long ga = (unsigned long long)(size_t)gsrc;
  g0[0] = 1u;                                   // count=1 (valid descriptor)
  g0[1] = ldsa;                                 // lds_addr
  g0[2] = (unsigned)ga;                         // global_addr[31:0]
  g0[3] = (unsigned)((ga >> 32) & 0x01FFFFFFull) | 0x80000000u; // addr[56:32]|type=2
  g1[0] = (1 << 16)        // data_size = 2 bytes
        | (1 << 20)        // pad_enable
        | (5 << 22)        // pad_interval: 64 DWORDs (one 128-half row)
        | (3 << 25);       // pad_amount: 4 DWORDs (8 halves)
  g1[1] = (int)(512u << 16);   // tensor_dim0 = 512 (low 16 in [31:16])
  g1[2] = (int)(64u << 16);    // tensor_dim0 hi=0; tensor_dim1 = 64
  g1[3] = (int)(128u << 16);   // tensor_dim1 hi=0; tile_dim0 = 128
  g1[4] = 64;                  // tile_dim1 = 64; tile_dim2 = 0
  g1[5] = 512;                 // tensor_dim0_stride = 512 (low 32)
  g1[6] = 0; g1[7] = 0;
#if __clang_major__ >= 23
  i32x8 z8 = {};
  __builtin_amdgcn_tensor_load_to_lds(g0, g1, z4, z4, z8, 0);
#else
  __builtin_amdgcn_tensor_load_to_lds(g0, g1, z4, z4, 0);
#endif
}

// ---------------------------------------------------------------------------
// Generic GEMM: C(M,N) = act(A(M,K) @ W(K,N) + bias)
// A_LAYOUT 0: A row-major (M,K); 1: A is (B,H,L,E) viewed as (B*L, H*E).
// ACT 0: id, 1: relu, 2: tanh*pi.  (compile-time -> branch-free staging)
// Block 256 thr = 8 waves; 64x64 block tile, K-step 32, LDS-staged f16 tiles.
// ---------------------------------------------------------------------------
template <int A_LAYOUT>
__device__ __forceinline__ const float* gemm_a_ptr(const float* __restrict__ A,
                                                   int m, int kk, int K) {
  if (A_LAYOUT == 0) return A + (size_t)m * K + kk;
  int b = m >> 9, l = m & 511;
  int h = kk >> 6, e = kk & 63;
  return A + (((size_t)b * HH + h) * LL + l) * EE + e;
}

template <int A_LAYOUT, int ACT>
__global__ void wmma_gemm_act(const float* __restrict__ A,
                              const float* __restrict__ W,
                              const float* __restrict__ bias,
                              float* __restrict__ C,
                              int M, int N, int K) {
  __shared__ _Float16 sA[64][40];   // [row][k], padded
  __shared__ _Float16 sW[64][40];   // transposed: [n][k]
  int nb   = (N + 63) >> 6;
  int bidn = blockIdx.x % nb;
  int bidm = blockIdx.x / nb;
  int bm = bidm * 64, bn = bidn * 64;
  int tid = threadIdx.x;
  int w = tid >> 5, lane = tid & 31;
  int wr = w & 3, wc = w >> 2;
  int hs = (lane >> 4) & 1, ln = lane & 15;
  v8f acc0 = {}; v8f acc1 = {};
  for (int kk = 0; kk < K; kk += 32) {
    // Stage A: 64 rows x 32 k, float4 loads -> ds_store_b64
#pragma unroll
    for (int i = tid; i < 512; i += 256) {
      int r = i >> 3, kc4 = (i & 7) * 4;
      float4 x = *(const float4*)gemm_a_ptr<A_LAYOUT>(A, bm + r, kk + kc4, K);
      *(v4h*)&sA[r][kc4] = cvt4(x);
    }
    // Stage W transposed: 32 k x 64 n, float4 loads -> 4 scattered b16 stores
#pragma unroll
    for (int i = tid; i < 512; i += 256) {
      int n4 = (i & 15) * 4, kc = i >> 4;
      int gn = bn + n4;
      float4 x = make_float4(0.f, 0.f, 0.f, 0.f);
      if (gn < N) x = *(const float4*)&W[(size_t)(kk + kc) * N + gn];
      sW[n4 + 0][kc] = (_Float16)x.x;
      sW[n4 + 1][kc] = (_Float16)x.y;
      sW[n4 + 2][kc] = (_Float16)x.z;
      sW[n4 + 3][kc] = (_Float16)x.w;
    }
    // Software prefetch of the next K-chunk (global_prefetch_b8 path)
    if (kk + 32 < K) {
      if (tid < 64) {
        __builtin_prefetch(gemm_a_ptr<A_LAYOUT>(A, bm + tid, kk + 32, K), 0, 3);
      } else if (tid < 96 && bn < N) {
        __builtin_prefetch(&W[(size_t)(kk + 32 + (tid - 64)) * N + bn], 0, 3);
      }
    }
    __syncthreads();
    v16h a  = afrag_f16(&sA[wr * 16 + ln][0], hs);
    v16h b0 = *(const v16h*)&sW[wc * 32 + ln][hs * 16];
    v16h b1 = *(const v16h*)&sW[wc * 32 + 16 + ln][hs * 16];
    acc0 = wmma_f16(a, b0, acc0);
    acc1 = wmma_f16(a, b1, acc1);
    __syncthreads();
  }
  int col0 = bn + wc * 32 + ln;
  int col1 = col0 + 16;
  int rowb = bm + wr * 16 + 8 * hs;
  float bv0 = (col0 < N) ? bias[col0] : 0.f;
  float bv1 = (col1 < N) ? bias[col1] : 0.f;
#pragma unroll
  for (int r = 0; r < 8; ++r) {
    int row = rowb + r;
    float v0 = acc0[r] + bv0;
    float v1 = acc1[r] + bv1;
    if (ACT == 1)      { v0 = fmaxf(v0, 0.f);   v1 = fmaxf(v1, 0.f); }
    else if (ACT == 2) { v0 = tanhf(v0) * PI_F; v1 = tanhf(v1) * PI_F; }
    if (col0 < N) C[(size_t)row * N + col0] = v0;
    if (col1 < N) C[(size_t)row * N + col1] = v1;
  }
}

// ---------------------------------------------------------------------------
// Series decomposition: sliding-window mean (K=25, replicate pad).
// One thread per (b, h*E+e) column; sequential over L.
// MODE 0 (q/k): residual f32 -> (B,L,H,E).
// MODE 1 (v):   residual f16 -> (B,H,E,L); trend f32 -> (B,H,L,E) and
//               trend f16 -> (B,H,E,L)  (TDM-ready transposed layouts).
// ---------------------------------------------------------------------------
template <int MODE>
__global__ void decomp_kernel(const float* __restrict__ X,
                              float* __restrict__ Rres,
                              float* __restrict__ Rtrend,
                              _Float16* __restrict__ RresH,
                              _Float16* __restrict__ RtrendH) {
  int t = blockIdx.x * 256 + threadIdx.x;   // 0..B*512-1
  int b = t >> 9, he = t & 511;
  int h = he >> 6, e = he & 63;
  const float* x = X + (size_t)b * LL * DM + he;  // stride DM over l
  float wsum = 12.f * x[0];
  for (int i = 0; i <= 12; ++i) wsum += x[(size_t)i * DM];
  const float inv = 1.f / 25.f;
  size_t tco = (((size_t)b * HH + h) * EE + e) * LL;  // (B,H,E,L) column base
  for (int l = 0; l < LL; ++l) {
    float xc = x[(size_t)l * DM];
    float ma = wsum * inv;
    if (MODE == 0) {
      Rres[((size_t)b * LL + l) * DM + he] = xc - ma;
    } else {
      RresH[tco + l]   = (_Float16)(xc - ma);
      RtrendH[tco + l] = (_Float16)ma;
      Rtrend[(((size_t)b * HH + h) * LL + l) * EE + e] = ma;
    }
    int lo = l - 12; if (lo < 0) lo = 0;
    int hi = l + 13; if (hi > LL - 1) hi = LL - 1;
    wsum += x[(size_t)hi * DM] - x[(size_t)lo * DM];
  }
}

// ---------------------------------------------------------------------------
// Rotation: out[b,h,l, m*128 + {e | 64+e}] = x*cos(phase_m) | x*sin(phase_m)
// res: (B,L,H,E) f32, om/th: (B,L,H,2) f32, out: (B,H,L,256) f16.
// ---------------------------------------------------------------------------
__global__ void rotate_kernel(const float* __restrict__ res,
                              const float* __restrict__ om,
                              const float* __restrict__ th,
                              _Float16* __restrict__ out) {
  size_t t = (size_t)blockIdx.x * 256 + threadIdx.x;  // B*L*H*E threads
  int e = (int)(t & 63); size_t r = t >> 6;
  int h = (int)(r & 7);  r >>= 3;
  int l = (int)(r & 511);
  int b = (int)(r >> 9);
  float x = res[t];
  size_t po = (((size_t)b * LL + l) * HH + h) * 2;
  float lf = (float)l;
  float p0 = om[po]     * lf + th[po];
  float p1 = om[po + 1] * lf + th[po + 1];
  float c0, s0, c1, s1;
  __sincosf(p0, &s0, &c0);
  __sincosf(p1, &s1, &c1);
  size_t ob = ((((size_t)b * HH + h) * LL + l) * RD) + e;
  out[ob]       = (_Float16)(x * c0);
  out[ob + 64]  = (_Float16)(x * s0);
  out[ob + 128] = (_Float16)(x * c1);
  out[ob + 192] = (_Float16)(x * s1);
}

// ---------------------------------------------------------------------------
// Scores: attn_raw[bh,l,s] = (qr[bh,l,:] . kr[bh,s,:]) / 8.
// One wave per 32x64 strip (2 row-tiles x 4 col-tiles): per K-step
// 2 A-frag + 4 B-frag loads feed 8 WMMAs (fragment register reuse).
// All fragments straight from global f16; qr+kr (67MB) lives in the 192MB L2.
// ---------------------------------------------------------------------------
__global__ void scores_kernel(const _Float16* __restrict__ qr,
                              const _Float16* __restrict__ kr,
                              float* __restrict__ attn) {
  int w = threadIdx.x >> 5, lane = threadIdx.x & 31;
  int gw = blockIdx.x * 8 + w;        // 128 jobs per (b,h)
  int bh = gw >> 7;
  int job = gw & 127;
  int tr = job >> 3;                  // row pair   -> rows tr*32
  int tc = job & 7;                   // col group  -> cols tc*64
  int hs = (lane >> 4) & 1, ln = lane & 15;
  const _Float16* Q  = qr + (size_t)bh * LL * RD;
  const _Float16* Kr = kr + (size_t)bh * LL * RD;
  v8f zero = {};
  v8f acc[2][4];
#pragma unroll
  for (int i = 0; i < 2; ++i)
#pragma unroll
    for (int j = 0; j < 4; ++j) acc[i][j] = zero;
#pragma unroll
  for (int k = 0; k < RD; k += 32) {
    v16h a0 = afrag_f16(&Q[(size_t)(tr * 32 + ln) * RD + k], hs);
    v16h a1 = afrag_f16(&Q[(size_t)(tr * 32 + 16 + ln) * RD + k], hs);
#pragma unroll
    for (int cj = 0; cj < 4; ++cj) {
      v16h b = *(const v16h*)&Kr[(size_t)(tc * 64 + cj * 16 + ln) * RD + k + hs * 16];
      acc[0][cj] = wmma_f16(a0, b, acc[0][cj]);
      acc[1][cj] = wmma_f16(a1, b, acc[1][cj]);
    }
  }
  float* O = attn + (size_t)bh * LL * LL;
  const float sc = 0.125f;   // 1/sqrt(64)
#pragma unroll
  for (int rt = 0; rt < 2; ++rt)
#pragma unroll
    for (int cj = 0; cj < 4; ++cj)
#pragma unroll
      for (int r = 0; r < 8; ++r) {
        int row = tr * 32 + rt * 16 + r + 8 * hs;
        O[(size_t)row * LL + tc * 64 + cj * 16 + ln] = acc[rt][cj][r] * sc;
      }
}

// ---------------------------------------------------------------------------
// Row softmax in place; also emits an f16 mirror for the attn@v WMMA GEMM.
// ---------------------------------------------------------------------------
__global__ void softmax_kernel(float* __restrict__ attn,
                               _Float16* __restrict__ attn_h) {
  __shared__ float red[256];
  float* p = attn + (size_t)blockIdx.x * LL;
  _Float16* ph = attn_h + (size_t)blockIdx.x * LL;
  int tid = threadIdx.x;
  float a = p[tid], b = p[tid + 256];
  red[tid] = fmaxf(a, b);
  __syncthreads();
  for (int s = 128; s > 0; s >>= 1) {
    if (tid < s) red[tid] = fmaxf(red[tid], red[tid + s]);
    __syncthreads();
  }
  float m = red[0];
  __syncthreads();
  float ea = __expf(a - m), eb = __expf(b - m);
  red[tid] = ea + eb;
  __syncthreads();
  for (int s = 128; s > 0; s >>= 1) {
    if (tid < s) red[tid] += red[tid + s];
    __syncthreads();
  }
  float inv = 1.f / red[0];
  float ra = ea * inv, rb = eb * inv;
  p[tid] = ra;               p[tid + 256] = rb;
  ph[tid] = (_Float16)ra;    ph[tid + 256] = (_Float16)rb;
}

// ---------------------------------------------------------------------------
// Per-(b,h) GEMM: Y[bh,i,e] = sum_l A[i,l] * X[bh,l,e]
// A: f16 512-wide rows (per-bh attn_h, or shared Uh/Uth).
// X: f16 transposed (B,H,E,L); each 64x128 K-chunk staged to LDS by the TDM.
// OUTMODE 0: Yf = acc (f32, (B,H,L,E));  1: Yf += acc;
//         2: Yh = f16 transposed (B,H,E,L) with rowscale folded in.
// ---------------------------------------------------------------------------
template <bool PER_BH, int OUTMODE>
__global__ void bh_gemm_kernel(const _Float16* __restrict__ Ah,
                               const _Float16* __restrict__ Xt,
                               const float* __restrict__ rowscale,
                               float* __restrict__ Yf,
                               _Float16* __restrict__ Yh) {
  __shared__ _Float16 sXt[64][136];   // [e][k-chunk of 128], TDM-padded rows
  int bh = blockIdx.x >> 1;
  int rb = blockIdx.x & 1;            // 256-row half
  const _Float16* A  = PER_BH ? (Ah + (size_t)bh * LL * LL) : Ah;
  const _Float16* Xb = Xt + (size_t)bh * EE * LL;
  int tid = threadIdx.x, w = tid >> 5, lane = tid & 31;
  int hs = (lane >> 4) & 1, ln = lane & 15;
  v8f zero = {};
  v8f acc[2][4];
#pragma unroll
  for (int i = 0; i < 2; ++i)
#pragma unroll
    for (int j = 0; j < 4; ++j) acc[i][j] = zero;

  for (int kb = 0; kb < LL; kb += 128) {
    // Wave 0 drives the Tensor Data Mover; everyone else just hits the barrier.
    if (tid < 32) {
      tdm_stage_64x128(&sXt[0][0], Xb + kb);
      __builtin_amdgcn_s_wait_tensorcnt((short)0);
    }
    __syncthreads();
#pragma unroll
    for (int ks = 0; ks < 4; ++ks) {
      int k = kb + ks * 32;
#pragma unroll
      for (int rt = 0; rt < 2; ++rt) {
        int arow = rb * 256 + (w * 2 + rt) * 16 + ln;
        v16h a = afrag_f16(&A[(size_t)arow * LL + k], hs);
#pragma unroll
        for (int cj = 0; cj < 4; ++cj) {
          v16h b = *(const v16h*)&sXt[cj * 16 + ln][ks * 32 + hs * 16];
          acc[rt][cj] = wmma_f16(a, b, acc[rt][cj]);
        }
      }
    }
    __syncthreads();
  }
#pragma unroll
  for (int rt = 0; rt < 2; ++rt)
#pragma unroll
    for (int cj = 0; cj < 4; ++cj)
#pragma unroll
      for (int r = 0; r < 8; ++r) {
        int row = rb * 256 + (w * 2 + rt) * 16 + r + 8 * hs;
        int col = cj * 16 + ln;
        float v = acc[rt][cj][r];
        if (OUTMODE == 0) {
          Yf[((size_t)bh * LL + row) * EE + col] = v;
        } else if (OUTMODE == 1) {
          size_t idx = ((size_t)bh * LL + row) * EE + col;
          Yf[idx] = Yf[idx] + v;
        } else {
          float s = rowscale[(size_t)bh * LL + row];
          Yh[((size_t)bh * EE + col) * LL + row] = (_Float16)(v * s);
        }
      }
}

// ---------------------------------------------------------------------------
// lambda conv + weights: w[bh,l] = 1/(1 + (1+elu(conv+blam)) * S[l])
// ---------------------------------------------------------------------------
__global__ void lambda_kernel(const float* __restrict__ vtrend,
                              const float* __restrict__ Wlam,
                              const float* __restrict__ blam,
                              const float* __restrict__ S,
                              float* __restrict__ wout) {
  int t = blockIdx.x * 256 + threadIdx.x;   // B*H*L threads
  int l = t & 511, bh = t >> 9;
  const float* V = vtrend + (size_t)bh * LL * EE;
  float acc = blam[0];
#pragma unroll
  for (int tt = 0; tt < 7; ++tt) {
    int ll = l + tt - 3;
    ll = ll < 0 ? 0 : (ll > LL - 1 ? LL - 1 : ll);
    const float4* vp = (const float4*)(V + (size_t)ll * EE);
#pragma unroll
    for (int e4 = 0; e4 < 16; ++e4) {
      float4 vv = vp[e4];
      acc += Wlam[(e4 * 4 + 0) * 7 + tt] * vv.x
           + Wlam[(e4 * 4 + 1) * 7 + tt] * vv.y
           + Wlam[(e4 * 4 + 2) * 7 + tt] * vv.z
           + Wlam[(e4 * 4 + 3) * 7 + tt] * vv.w;
    }
  }
  float elu = acc > 0.f ? acc : (__expf(acc) - 1.f);
  float lam = 1.f + elu;
  wout[t] = 1.f / (1.f + lam * S[l]);
}

// Convert U to f16 (Uh = U, Uth = U^T), feeding the trend-refinement GEMMs.
__global__ void prep_U(const float* __restrict__ U,
                       _Float16* __restrict__ Uh,
                       _Float16* __restrict__ Uth) {
  int t = blockIdx.x * 256 + threadIdx.x;   // 512*512 threads
  int c = t & 511, r = t >> 9;
  float x = U[t];
  Uh[t] = (_Float16)x;
  Uth[(size_t)c * LL + r] = (_Float16)x;
}

// ---------------------------------------------------------------------------
// Penalties: om_pen = sum (first-diff of q_om,k_om)^2; th_pen = sum sq(q_th,k_th)
// ---------------------------------------------------------------------------
__global__ void zero2_kernel(float* a, float* b) { *a = 0.f; *b = 0.f; }

__global__ void penalty_kernel(const float* __restrict__ qom,
                               const float* __restrict__ kom,
                               const float* __restrict__ qth,
                               const float* __restrict__ kth,
                               float* __restrict__ om_pen,
                               float* __restrict__ th_pen) {
  __shared__ float s1[256], s2[256];
  int t = blockIdx.x * 256 + threadIdx.x;   // B*L*H*M = 131072
  float o = 0.f, p = 0.f;
  float a = qth[t], c = kth[t];
  p = a * a + c * c;
  int l = (t >> 4) & 511;
  if (l < LL - 1) {
    float d1 = qom[t + 16] - qom[t];
    float d2 = kom[t + 16] - kom[t];
    o = d1 * d1 + d2 * d2;
  }
  s1[threadIdx.x] = o; s2[threadIdx.x] = p;
  __syncthreads();
  for (int s = 128; s > 0; s >>= 1) {
    if (threadIdx.x < s) {
      s1[threadIdx.x] += s1[threadIdx.x + s];
      s2[threadIdx.x] += s2[threadIdx.x + s];
    }
    __syncthreads();
  }
  if (threadIdx.x == 0) {
    atomicAdd(om_pen, s1[0]);
    atomicAdd(th_pen, s2[0]);
  }
}

// ---------------------------------------------------------------------------
extern "C" void kernel_launch(void* const* d_in, const int* in_sizes, int n_in,
                              void* d_out, int out_size, void* d_ws, size_t ws_size,
                              hipStream_t stream) {
  const float* queries = (const float*)d_in[0];
  const float* keys    = (const float*)d_in[1];
  const float* values  = (const float*)d_in[2];
  const float* Wq  = (const float*)d_in[3];  const float* bq  = (const float*)d_in[4];
  const float* Wk  = (const float*)d_in[5];  const float* bk  = (const float*)d_in[6];
  const float* Wv  = (const float*)d_in[7];  const float* bv  = (const float*)d_in[8];
  const float* Wqo = (const float*)d_in[9];  const float* bqo = (const float*)d_in[10];
  const float* Wko = (const float*)d_in[11]; const float* bko = (const float*)d_in[12];
  const float* Wqt = (const float*)d_in[13]; const float* bqt = (const float*)d_in[14];
  const float* Wkt = (const float*)d_in[15]; const float* bkt = (const float*)d_in[16];
  const float* Wo  = (const float*)d_in[17]; const float* bo  = (const float*)d_in[18];
  const float* Wlam = (const float*)d_in[19]; const float* blam = (const float*)d_in[20];
  const float* U   = (const float*)d_in[21]; const float* S   = (const float*)d_in[22];

  float* out_main = (float*)d_out;                               // (B,L,512)
  float* attn     = out_main + (size_t)BB * LL * DM;             // (B,H,L,L)
  float* om_pen   = attn + (size_t)BB * HH * LL * LL;
  float* th_pen   = om_pen + 1;

  char* wsp = (char*)d_ws;
  size_t off = 0;
  auto alloc = [&](size_t bytes) -> void* {
    void* p = wsp + off;
    off = (off + bytes + 255) & ~(size_t)255;
    return p;
  };

  const size_t BLD  = (size_t)BB * LL * DM;     // 4.19M elems
  const size_t BLHM = (size_t)BB * LL * HH * 2; // 131072
  float* q     = (float*)alloc(BLD * 4);
  float* k     = (float*)alloc(BLD * 4);
  float* v     = (float*)alloc(BLD * 4);
  float* qom   = (float*)alloc(BLHM * 4);
  float* kom   = (float*)alloc(BLHM * 4);
  float* qth   = (float*)alloc(BLHM * 4);
  float* kth   = (float*)alloc(BLHM * 4);
  float* qres  = (float*)alloc(BLD * 4);        // (B,L,H,E)
  float* kres  = (float*)alloc(BLD * 4);
  float* vtr   = (float*)alloc(BLD * 4);        // (B,H,L,E) f32 (lambda conv)
  _Float16* vresh = (_Float16*)alloc(BLD * 2);  // (B,H,E,L) f16
  _Float16* vtrh  = (_Float16*)alloc(BLD * 2);  // (B,H,E,L) f16
  _Float16* qr = (_Float16*)alloc((size_t)BB * HH * LL * RD * 2);
  _Float16* kr = (_Float16*)alloc((size_t)BB * HH * LL * RD * 2);
  _Float16* attn_h = (_Float16*)alloc((size_t)BB * HH * LL * LL * 2);
  float* outpre = (float*)alloc(BLD * 4);       // (B,H,L,E)
  _Float16* t1h = (_Float16*)alloc(BLD * 2);    // (B,H,E,L) f16, w-scaled
  float* wvec  = (float*)alloc((size_t)BB * HH * LL * 4);
  _Float16* Uh  = (_Float16*)alloc((size_t)LL * LL * 2);
  _Float16* Uth = (_Float16*)alloc((size_t)LL * LL * 2);

  // 1) Projections (WMMA GEMMs)
  wmma_gemm_act<0, 0><<<128 * 8, 256, 0, stream>>>(queries, Wq, bq, q, 8192, 512, 512);
  wmma_gemm_act<0, 0><<<128 * 8, 256, 0, stream>>>(keys,    Wk, bk, k, 8192, 512, 512);
  wmma_gemm_act<0, 0><<<128 * 8, 256, 0, stream>>>(values,  Wv, bv, v, 8192, 512, 512);
  wmma_gemm_act<0, 1><<<128, 256, 0, stream>>>(queries, Wqo, bqo, qom, 8192, 16, 512);
  wmma_gemm_act<0, 1><<<128, 256, 0, stream>>>(keys,    Wko, bko, kom, 8192, 16, 512);
  wmma_gemm_act<0, 2><<<128, 256, 0, stream>>>(queries, Wqt, bqt, qth, 8192, 16, 512);
  wmma_gemm_act<0, 2><<<128, 256, 0, stream>>>(keys,    Wkt, bkt, kth, 8192, 16, 512);

  // 2) Series decomposition
  decomp_kernel<0><<<32, 256, 0, stream>>>(q, qres, nullptr, nullptr, nullptr);
  decomp_kernel<0><<<32, 256, 0, stream>>>(k, kres, nullptr, nullptr, nullptr);
  decomp_kernel<1><<<32, 256, 0, stream>>>(v, nullptr, vtr, vresh, vtrh);

  // 3) Periodic rotation -> f16 (B,H,L,256)
  rotate_kernel<<<16384, 256, 0, stream>>>(qres, qom, qth, qr);
  rotate_kernel<<<16384, 256, 0, stream>>>(kres, kom, kth, kr);

  // 4) Attention scores (WMMA) straight into d_out's attn region; softmax in
  //    place + f16 mirror for the next GEMM.
  scores_kernel<<<2048, 256, 0, stream>>>(qr, kr, attn);
  softmax_kernel<<<BB * HH * LL, 256, 0, stream>>>(attn, attn_h);

  // 5) out_pre = attn @ v_res (WMMA; X tiles DMA'd to LDS by the TDM)
  bh_gemm_kernel<true, 0><<<BB * HH * 2, 256, 0, stream>>>(attn_h, vresh, nullptr, outpre, nullptr);

  // 6) Trend refinement: out_pre += U @ (w .* (U^T @ v_trend))
  lambda_kernel<<<256, 256, 0, stream>>>(vtr, Wlam, blam, S, wvec);
  prep_U<<<1024, 256, 0, stream>>>(U, Uh, Uth);
  bh_gemm_kernel<false, 2><<<BB * HH * 2, 256, 0, stream>>>(Uth, vtrh, wvec, nullptr, t1h);
  bh_gemm_kernel<false, 1><<<BB * HH * 2, 256, 0, stream>>>(Uh, t1h, nullptr, outpre, nullptr);

  // 7) Output projection (WMMA, A in (B,H,L,E) layout)
  wmma_gemm_act<1, 0><<<128 * 8, 256, 0, stream>>>(outpre, Wo, bo, out_main, 8192, 512, 512);

  // 8) Smoothness penalties
  zero2_kernel<<<1, 1, 0, stream>>>(om_pen, th_pen);
  penalty_kernel<<<512, 256, 0, stream>>>(qom, kom, qth, kth, om_pen, th_pen);
}
